// Trans_14310831030699
// MI455X (gfx1250) — matrix operand order repaired and checked by reading, not compiled, not run on hardware
//
#include <hip/hip_runtime.h>
#include <hip/hip_bf16.h>
#include <math.h>

// Problem dims (compile-time constants from the reference)
#define B_   2
#define S_   1024
#define D_   1024
#define H_   16
#define L_   6
#define FF_  4096
#define V_   32000
#define DK_  64
#define EPS_ 1e-6f

typedef __attribute__((ext_vector_type(16))) __bf16 v16bf;
typedef __attribute__((ext_vector_type(8)))  float  v8f;

// ---------------- GEMM: C = act(alpha*(A@B) + bias [+ resid]) ----------------
// A: f32 [M,K] lda ; B: f32 [K,N] ldb (or [N,K] if transB) ; C: f32 [M,N] ldc
// All launched shapes are exact multiples: M%64==0, N%64==0, K%32==0 -> no guards.
// bf16 WMMA 16x16x32, 64x64 block tile, BK=32, 128 threads (4 waves).
#define BM 64
#define BN 64
#define BK 32
#define LDS_STRIDE (BK + 8)   // 40 bf16 = 80 B row stride: 16B-aligned, bank-spread

__global__ void __launch_bounds__(128) gemm_bf16_wmma(
    const float* __restrict__ A, int lda,
    const float* __restrict__ Bm, int ldb, int transB,
    const float* __restrict__ bias,
    const float* __restrict__ resid,
    float* __restrict__ C, int ldc,
    int M, int N, int K, float alpha, int relu)
{
  __shared__ __bf16 As [BM][LDS_STRIDE];   // As[m][k]
  __shared__ __bf16 BsT[BN][LDS_STRIDE];   // BsT[n][k] (B stored transposed)

  const int tid  = threadIdx.x;
  const int wave = tid >> 5;
  const int lane = tid & 31;
  const int m0 = blockIdx.y * BM;
  const int n0 = blockIdx.x * BN;
  const int wr = (wave >> 1) * 32;   // wave row offset in tile
  const int wc = (wave & 1) * 32;    // wave col offset in tile
  const int hi = lane >> 4;          // half-wave: 0 or 1
  const int lr = lane & 15;
  const int kbA = hi * 16;           // A frag K base: lanes16-31 start at K=16

  v8f acc[2][2] = {};

  for (int k0 = 0; k0 < K; k0 += BK) {
    // ---- Stage A tile: 64x32 f32 -> bf16, vectorized float4 loads ----
#pragma unroll
    for (int p = 0; p < 4; ++p) {
      int idx = p * 128 + tid;          // float4 index within tile
      int r   = idx >> 3;               // BK/4 = 8 float4 per row
      int c4  = (idx & 7) << 2;
      const float4 f = *(const float4*)(A + (size_t)(m0 + r) * lda + k0 + c4);
      __bf16* dst = &As[r][c4];
      dst[0] = (__bf16)f.x; dst[1] = (__bf16)f.y;
      dst[2] = (__bf16)f.z; dst[3] = (__bf16)f.w;
    }
    // ---- Stage B tile transposed into BsT[n][k] ----
    if (transB) {
      // B is [N,K] row-major: contiguous along k -> contiguous LDS row writes
#pragma unroll
      for (int p = 0; p < 4; ++p) {
        int idx = p * 128 + tid;
        int n   = idx >> 3;             // 8 float4 per n row
        int k4  = (idx & 7) << 2;
        const float4 f = *(const float4*)(Bm + (size_t)(n0 + n) * ldb + k0 + k4);
        __bf16* dst = &BsT[n][k4];
        dst[0] = (__bf16)f.x; dst[1] = (__bf16)f.y;
        dst[2] = (__bf16)f.z; dst[3] = (__bf16)f.w;
      }
    } else {
      // B is [K,N] row-major: contiguous along n -> scatter into BsT columns
#pragma unroll
      for (int p = 0; p < 4; ++p) {
        int idx = p * 128 + tid;
        int r   = idx >> 4;             // BN/4 = 16 float4 per k row
        int c4  = (idx & 15) << 2;
        const float4 f = *(const float4*)(Bm + (size_t)(k0 + r) * ldb + n0 + c4);
        BsT[c4 + 0][r] = (__bf16)f.x;
        BsT[c4 + 1][r] = (__bf16)f.y;
        BsT[c4 + 2][r] = (__bf16)f.z;
        BsT[c4 + 3][r] = (__bf16)f.w;
      }
    }

    // Prefetch next K-tile while this one is consumed (global_prefetch_b8)
    if (k0 + BK < K) {
      __builtin_prefetch(A + (size_t)(m0 + (tid & 63)) * lda + k0 + BK, 0, 1);
      const float* bp = transB ? (Bm + (size_t)(n0 + (tid & 63)) * ldb + k0 + BK)
                               : (Bm + (size_t)(k0 + BK + (tid & 31)) * ldb + n0);
      __builtin_prefetch(bp, 0, 1);
    }

    __syncthreads();

    // ---- Load fragments (contiguous k = 16*hi + s for A and B), contiguous LDS reads ----
    v16bf af[2], bfr[2];
#pragma unroll
    for (int i = 0; i < 2; ++i) {
      const __bf16* ap = &As[wr + i * 16 + lr][0];
#pragma unroll
      for (int j = 0; j < 8; ++j) {
        af[i][j]     = ap[kbA + j];          // K = kbA..kbA+7
        af[i][j + 8] = ap[kbA + 8 + j];      // K = kbA+8..kbA+15
      }
      const __bf16* bp = &BsT[wc + i * 16 + lr][hi * 16];  // 16 contiguous bf16
#pragma unroll
      for (int j = 0; j < 16; ++j) bfr[i][j] = bp[j];
    }

#pragma unroll
    for (int i = 0; i < 2; ++i)
#pragma unroll
      for (int j = 0; j < 2; ++j)
        acc[i][j] = __builtin_amdgcn_wmma_f32_16x16x32_bf16(
            false, af[i], false, bfr[j], (short)0, acc[i][j], false, false);

    __syncthreads();
  }

  // ---- Store: C/D layout — VGPR r: M = r + 8*(lane>=16), N = lane&15 ----
#pragma unroll
  for (int i = 0; i < 2; ++i)
#pragma unroll
    for (int j = 0; j < 2; ++j)
#pragma unroll
      for (int r = 0; r < 8; ++r) {
        int gm = m0 + wr + i * 16 + r + hi * 8;
        int gn = n0 + wc + j * 16 + lr;
        float v = acc[i][j][r] * alpha;
        if (bias)  v += bias[gn];
        if (resid) v += resid[(size_t)gm * ldc + gn];
        if (relu)  v = fmaxf(v, 0.0f);
        C[(size_t)gm * ldc + gn] = v;
      }
}

// ---------------- Embedding + sinusoidal positional encoding ----------------
__global__ void embed_pe_kernel(const int* __restrict__ x,
                                const float* __restrict__ emb,
                                float* __restrict__ h)
{
  int bs = blockIdx.x;          // 0..B*S-1
  int s  = bs % S_;
  int tok = x[bs];
  const float scale = 32.0f;    // sqrt(D)=sqrt(1024)
  const float c = -9.2103403719761836f / (float)D_;  // -log(10000)/D
  for (int d = threadIdx.x; d < D_; d += blockDim.x) {
    float dt  = __expf(c * (float)(d & ~1));
    float ang = (float)s * dt;
    float pe  = (d & 1) ? __cosf(ang) : __sinf(ang);
    h[(size_t)bs * D_ + d] = emb[(size_t)tok * D_ + d] * scale + pe;
  }
}

// ---------------- LayerNorm (torch semantics: ddof=1, eps on std) -----------
__global__ void layernorm_kernel(const float* __restrict__ in,
                                 const float* __restrict__ g,
                                 const float* __restrict__ be,
                                 float* __restrict__ out)
{
  __shared__ float red[256];
  int row = blockIdx.x;
  int tid = threadIdx.x;
  const float* xr = in + (size_t)row * D_;

  float s = 0.0f;
  for (int d = tid; d < D_; d += 256) s += xr[d];
  red[tid] = s; __syncthreads();
  for (int off = 128; off > 0; off >>= 1) {
    if (tid < off) red[tid] += red[tid + off];
    __syncthreads();
  }
  float mean = red[0] / (float)D_;
  __syncthreads();

  float ss = 0.0f;
  for (int d = tid; d < D_; d += 256) { float t = xr[d] - mean; ss += t * t; }
  red[tid] = ss; __syncthreads();
  for (int off = 128; off > 0; off >>= 1) {
    if (tid < off) red[tid] += red[tid + off];
    __syncthreads();
  }
  float sd  = sqrtf(red[0] / (float)(D_ - 1));
  float inv = 1.0f / (sd + EPS_);
  for (int d = tid; d < D_; d += 256)
    out[(size_t)row * D_ + d] = g[d] * (xr[d] - mean) * inv + be[d];
}

// ---------------- Causal softmax over one [S,S] score matrix ----------------
__global__ void softmax_causal_kernel(float* __restrict__ sc)
{
  __shared__ float red[256];
  int qi  = blockIdx.x;
  int tid = threadIdx.x;
  float* row = sc + (size_t)qi * S_;
  int n = qi + 1;  // valid keys

  float mx = -1e30f;
  for (int j = tid; j < n; j += 256) mx = fmaxf(mx, row[j]);
  red[tid] = mx; __syncthreads();
  for (int off = 128; off > 0; off >>= 1) {
    if (tid < off) red[tid] = fmaxf(red[tid], red[tid + off]);
    __syncthreads();
  }
  mx = red[0]; __syncthreads();

  float s = 0.0f;
  for (int j = tid; j < n; j += 256) {
    float e = __expf(row[j] - mx);
    row[j] = e;
    s += e;
  }
  red[tid] = s; __syncthreads();
  for (int off = 128; off > 0; off >>= 1) {
    if (tid < off) red[tid] += red[tid + off];
    __syncthreads();
  }
  float inv = 1.0f / red[0];
  for (int j = tid; j < n; j += 256) row[j] *= inv;
  for (int j = n + tid; j < S_; j += 256) row[j] = 0.0f;  // masked tail
}

// ---------------------------------------------------------------------------
extern "C" void kernel_launch(void* const* d_in, const int* in_sizes, int n_in,
                              void* d_out, int out_size, void* d_ws, size_t ws_size,
                              hipStream_t stream)
{
  const int*   x    = (const int*)  d_in[0];
  const float* emb  = (const float*)d_in[1];
  const float* Wq   = (const float*)d_in[2];
  const float* bq   = (const float*)d_in[3];
  const float* Wk   = (const float*)d_in[4];
  const float* bk   = (const float*)d_in[5];
  const float* Wv   = (const float*)d_in[6];
  const float* bv   = (const float*)d_in[7];
  const float* Wo   = (const float*)d_in[8];
  const float* bo   = (const float*)d_in[9];
  const float* W1   = (const float*)d_in[10];
  const float* b1   = (const float*)d_in[11];
  const float* W2   = (const float*)d_in[12];
  const float* b2   = (const float*)d_in[13];
  const float* g1   = (const float*)d_in[14];
  const float* be1  = (const float*)d_in[15];
  const float* g2   = (const float*)d_in[16];
  const float* be2  = (const float*)d_in[17];
  const float* gf   = (const float*)d_in[18];
  const float* bef  = (const float*)d_in[19];
  const float* Wout = (const float*)d_in[20];
  const float* bout = (const float*)d_in[21];
  float* out = (float*)d_out;

  // Workspace layout (f32): ~84 MB
  float* ws = (float*)d_ws;
  const size_t BSD = (size_t)B_ * S_ * D_;
  float* h  = ws;
  float* hn = h  + BSD;
  float* q  = hn + BSD;
  float* k  = q  + BSD;
  float* v  = k  + BSD;
  float* ao = v  + BSD;
  float* sc = ao + BSD;                 // S*S
  float* ff = sc + (size_t)S_ * S_;     // B*S*FF

  const int BS = B_ * S_;
  dim3 blk(128);
  dim3 gDD(D_ / BN,  BS / BM);    // N=1024
  dim3 gF1(FF_ / BN, BS / BM);    // N=4096
  dim3 gVO(V_ / BN,  BS / BM);    // N=32000
  dim3 gSC(S_ / BN,  S_ / BM);    // scores 1024x1024
  dim3 gAO(DK_ / BN, S_ / BM);    // 1024x64

  embed_pe_kernel<<<BS, 256, 0, stream>>>(x, emb, h);

  for (int i = 0; i < L_; ++i) {
    const float* Wqi = Wq + (size_t)i * D_ * D_;
    const float* Wki = Wk + (size_t)i * D_ * D_;
    const float* Wvi = Wv + (size_t)i * D_ * D_;
    const float* Woi = Wo + (size_t)i * D_ * D_;
    const float* W1i = W1 + (size_t)i * D_ * FF_;
    const float* W2i = W2 + (size_t)i * FF_ * D_;

    // LN1 -> hn ; QKV projections
    layernorm_kernel<<<BS, 256, 0, stream>>>(h, g1 + i * D_, be1 + i * D_, hn);
    gemm_bf16_wmma<<<gDD, blk, 0, stream>>>(hn, D_, Wqi, D_, 0, bq + i * D_, nullptr, q, D_, BS, D_, D_, 1.0f, 0);
    gemm_bf16_wmma<<<gDD, blk, 0, stream>>>(hn, D_, Wki, D_, 0, bk + i * D_, nullptr, k, D_, BS, D_, D_, 1.0f, 0);
    gemm_bf16_wmma<<<gDD, blk, 0, stream>>>(hn, D_, Wvi, D_, 0, bv + i * D_, nullptr, v, D_, BS, D_, D_, 1.0f, 0);

    // Attention per (batch, head) reusing one S*S score buffer
    for (int b = 0; b < B_; ++b) {
      for (int hh = 0; hh < H_; ++hh) {
        const float* qh = q + (size_t)b * S_ * D_ + hh * DK_;
        const float* kh = k + (size_t)b * S_ * D_ + hh * DK_;
        const float* vh = v + (size_t)b * S_ * D_ + hh * DK_;
        float*      aoh = ao + (size_t)b * S_ * D_ + hh * DK_;
        // sc = (q @ k^T) / sqrt(DK)
        gemm_bf16_wmma<<<gSC, blk, 0, stream>>>(qh, D_, kh, D_, 1, nullptr, nullptr, sc, S_, S_, S_, DK_, 0.125f, 0);
        softmax_causal_kernel<<<S_, 256, 0, stream>>>(sc);
        // ao_h = aw @ v_h
        gemm_bf16_wmma<<<gAO, blk, 0, stream>>>(sc, S_, vh, D_, 0, nullptr, nullptr, aoh, D_, S_, DK_, S_, 1.0f, 0);
      }
    }

    // h = h + ao @ Wo + bo (fused residual)
    gemm_bf16_wmma<<<gDD, blk, 0, stream>>>(ao, D_, Woi, D_, 0, bo + i * D_, h, h, D_, BS, D_, D_, 1.0f, 0);

    // FF block: h = h + relu(LN2(h) @ W1 + b1) @ W2 + b2
    layernorm_kernel<<<BS, 256, 0, stream>>>(h, g2 + i * D_, be2 + i * D_, hn);
    gemm_bf16_wmma<<<gF1, blk, 0, stream>>>(hn, D_, W1i, FF_, 0, b1 + i * FF_, nullptr, ff, FF_, BS, FF_, D_, 1.0f, 1);
    gemm_bf16_wmma<<<gDD, blk, 0, stream>>>(ff, FF_, W2i, D_, 0, b2 + i * D_, h, h, D_, BS, D_, FF_, 1.0f, 0);
  }

  // Final LN + vocab projection
  layernorm_kernel<<<BS, 256, 0, stream>>>(h, gf, bef, hn);
  gemm_bf16_wmma<<<gVO, blk, 0, stream>>>(hn, D_, Wout, V_, 0, bout, nullptr, out, V_, BS, V_, D_, 1.0f, 0);

  (void)in_sizes; (void)n_in; (void)out_size; (void)ws_size;
}
